// RNN_35296041238973
// MI455X (gfx1250) — matrix operand order repaired and checked by reading, not compiled, not run on hardware
//
#include <hip/hip_runtime.h>
#include <hip/hip_bf16.h>
#include <stdint.h>

#define D 256
#define NEG_SLOPE 0.01f

typedef __attribute__((ext_vector_type(16))) __bf16 v16bf;
typedef __attribute__((ext_vector_type(8)))  float  v8f;

union ABFrag {
    unsigned u[8];
    uint4    q[2];
    v16bf    v;
};

// ---------- helpers ----------
static __device__ __forceinline__ unsigned short f2bf(float f) {
    unsigned u = __float_as_uint(f);
    u += 0x7FFFu + ((u >> 16) & 1u);          // round-to-nearest-even
    return (unsigned short)(u >> 16);
}
static __device__ __forceinline__ float bf2f(unsigned short h) {
    return __uint_as_float(((unsigned)h) << 16);
}
static __device__ __forceinline__ void atomAddF(float* p, float v) {
    __hip_atomic_fetch_add(p, v, __ATOMIC_RELAXED, __HIP_MEMORY_SCOPE_AGENT);
}

// ---------- kernels ----------

// x = feature + 0.8*hidden ; split into bf16 hi/lo, packed as (bf16 k, bf16 k+1) words
__global__ __launch_bounds__(256) void prep_kernel(
    const float* __restrict__ feat, const float* __restrict__ hid,
    unsigned* __restrict__ Ahi, unsigned* __restrict__ Alo, int nWords)
{
    int idx = blockIdx.x * 256 + threadIdx.x;
    if (idx >= nWords) return;
    int base = idx * 2;
    float s0 = feat[base]     + 0.8f * hid[base];
    float s1 = feat[base + 1] + 0.8f * hid[base + 1];
    unsigned short h0 = f2bf(s0), h1 = f2bf(s1);
    Ahi[idx] = (unsigned)h0 | ((unsigned)h1 << 16);
    unsigned short l0 = f2bf(s0 - bf2f(h0));
    unsigned short l1 = f2bf(s1 - bf2f(h1));
    Alo[idx] = (unsigned)l0 | ((unsigned)l1 << 16);
}

// Repack W (row-major DxD, row = K, col = N) into WMMA-B-fragment-ready layout:
// Wopt[(((tileN*8 + kstep)*32 + lane)*8 + j]  holds packed pair (K = kstep*32 + 2j + 16*(lane>=16),
// K+1) at column tileN*16 + (lane&15).  -> per-lane contiguous b128 loads, fully coalesced.
__global__ __launch_bounds__(256) void packW_kernel(
    const float* __restrict__ W, unsigned* __restrict__ Whi, unsigned* __restrict__ Wlo)
{
    int id = blockIdx.x * 256 + threadIdx.x;        // 16*8*32*8 = 32768
    if (id >= 32768) return;
    int j     =  id        & 7;
    int lane  = (id >> 3)  & 31;
    int kstep = (id >> 8)  & 7;
    int tileN =  id >> 11;
    int half = lane >> 4, lm = lane & 15;
    int k = kstep * 32 + 2 * j + 16 * half;
    int n = tileN * 16 + lm;
    float w0 = W[(size_t)k * D + n];
    float w1 = W[(size_t)(k + 1) * D + n];
    unsigned short h0 = f2bf(w0), h1 = f2bf(w1);
    Whi[id] = (unsigned)h0 | ((unsigned)h1 << 16);
    unsigned short l0 = f2bf(w0 - bf2f(h0));
    unsigned short l1 = f2bf(w1 - bf2f(h1));
    Wlo[id] = (unsigned)l0 | ((unsigned)l1 << 16);
}

// H = Xsum @ W via split-bf16 WMMA (3 products per K-step ~ fp32 accuracy).
// One wave per 16x16 output tile.
__global__ __launch_bounds__(256) void gemm_kernel(
    const unsigned* __restrict__ Ahi, const unsigned* __restrict__ Alo,
    const unsigned* __restrict__ Bhi, const unsigned* __restrict__ Blo,
    float* __restrict__ H, int nRows)
{
    int wave  = (blockIdx.x * 256 + threadIdx.x) >> 5;
    int lane  = threadIdx.x & 31;
    int tileM = wave >> 4;                 // 16 column tiles (D=256)
    int tileN = wave & 15;
    if (tileM * 16 >= nRows) return;       // wave-uniform guard (EXEC stays all-1s)
    int half = lane >> 4, lm = lane & 15;

    const unsigned* arow_hi = Ahi + (size_t)(tileM * 16 + lm) * (D / 2);
    const unsigned* arow_lo = Alo + (size_t)(tileM * 16 + lm) * (D / 2);

    v8f acc = {};
#pragma unroll
    for (int ks = 0; ks < 8; ++ks) {
        int abase = ks * 16 + 8 * half;    // kb/2 + 8*half
        ABFrag a_hi, a_lo, b_hi, b_lo;
        a_hi.q[0] = *(const uint4*)(arow_hi + abase);
        a_hi.q[1] = *(const uint4*)(arow_hi + abase + 4);
        a_lo.q[0] = *(const uint4*)(arow_lo + abase);
        a_lo.q[1] = *(const uint4*)(arow_lo + abase + 4);

        const unsigned* bph = Bhi + (size_t)(((tileN * 8 + ks) * 32 + lane)) * 8;
        const unsigned* bpl = Blo + (size_t)(((tileN * 8 + ks) * 32 + lane)) * 8;
        b_hi.q[0] = *(const uint4*)(bph);
        b_hi.q[1] = *(const uint4*)(bph + 4);
        b_lo.q[0] = *(const uint4*)(bpl);
        b_lo.q[1] = *(const uint4*)(bpl + 4);

        acc = __builtin_amdgcn_wmma_f32_16x16x32_bf16(false, a_hi.v, false, b_hi.v,
                                                      (short)0, acc, false, false);
        acc = __builtin_amdgcn_wmma_f32_16x16x32_bf16(false, a_hi.v, false, b_lo.v,
                                                      (short)0, acc, false, false);
        acc = __builtin_amdgcn_wmma_f32_16x16x32_bf16(false, a_lo.v, false, b_hi.v,
                                                      (short)0, acc, false, false);
    }

    float* out = H + (size_t)(tileM * 16) * D + tileN * 16;
#pragma unroll
    for (int i = 0; i < 8; ++i)
        out[(size_t)(i + 8 * half) * D + lm] = acc[i];
}

__global__ __launch_bounds__(256) void fill_kernel(float* __restrict__ p, float v, int n)
{
    int i = blockIdx.x * 256 + threadIdx.x;
    if (i < n) p[i] = v;
}

__global__ __launch_bounds__(256) void deg_kernel(
    const int* __restrict__ dst, float* __restrict__ deg, int nEdges)
{
    int e = blockIdx.x * 256 + threadIdx.x;
    if (e < nEdges) atomAddF(&deg[dst[e]], 1.0f);
}

__global__ __launch_bounds__(256) void dinv_kernel(float* __restrict__ deg, int n)
{
    int i = blockIdx.x * 256 + threadIdx.x;
    if (i >= n) return;
    float d = deg[i];
    deg[i] = d > 0.0f ? __frsqrt_rn(d) : 0.0f;
}

// out[dst] += H[src] * dinv[src]*dinv[dst] ; 64 threads per edge, 4 channels each.
__global__ __launch_bounds__(256) void scatter_kernel(
    const int* __restrict__ src, const int* __restrict__ dst,
    const float* __restrict__ dinv, const float* __restrict__ H,
    float* __restrict__ out, int nEdges)
{
    long long gid = (long long)blockIdx.x * 256 + threadIdx.x;
    int e = (int)(gid >> 6);
    if (e >= nEdges) return;
    int c = (int)(gid & 63) << 2;
    int s = src[e], d = dst[e];
    float w = dinv[s] * dinv[d];
    const float4 hv = *(const float4*)(H + (size_t)s * D + c);
    float* o = out + (size_t)d * D + c;
    atomAddF(o + 0, hv.x * w);
    atomAddF(o + 1, hv.y * w);
    atomAddF(o + 2, hv.z * w);
    atomAddF(o + 3, hv.w * w);
}

// out = leaky_relu(out_agg + H[i]*dinv[i]^2 + b)
__global__ __launch_bounds__(256) void epilogue_kernel(
    const float* __restrict__ H, const float* __restrict__ dinv,
    const float* __restrict__ b, float* __restrict__ out, long long total)
{
    long long idx = (long long)blockIdx.x * 256 + threadIdx.x;
    if (idx >= total) return;
    int j = (int)(idx & (D - 1));
    int i = (int)(idx >> 8);
    float di = dinv[i];
    float v = out[idx] + H[idx] * di * di + b[j];
    out[idx] = v > 0.0f ? v : v * NEG_SLOPE;
}

// ---------- launch ----------
extern "C" void kernel_launch(void* const* d_in, const int* in_sizes, int n_in,
                              void* d_out, int out_size, void* d_ws, size_t ws_size,
                              hipStream_t stream)
{
    const int nNodes = in_sizes[2] / D;         // 50000
    const int nEdges = in_sizes[0] / 2;         // 800000
    const size_t ND  = (size_t)nNodes * D;

    const float* W = (const float*)d_in[6];
    const float* b = (const float*)d_in[7];

    // workspace layout (bytes, 256-aligned sizes)
    char*  ws  = (char*)d_ws;
    size_t off = 0;
    unsigned* Ahi = (unsigned*)(ws + off); off += ND * 2;              // bf16-pair words
    unsigned* Alo = (unsigned*)(ws + off); off += ND * 2;
    float*    H   = (float*)   (ws + off); off += ND * 4;
    unsigned* Whi = (unsigned*)(ws + off); off += 32768 * 4;
    unsigned* Wlo = (unsigned*)(ws + off); off += 32768 * 4;
    float*    deg = (float*)   (ws + off); off += ((size_t)nNodes * 4 + 255) & ~(size_t)255;
    (void)ws_size;

    packW_kernel<<<32768 / 256, 256, 0, stream>>>(W, Whi, Wlo);

    const int prepBlocks  = (int)((ND / 2 + 255) / 256);
    const int gemmBlocks  = ((nNodes / 16) * 16 + 7) / 8;   // waves = rowTiles*16, 8 waves/block
    const int nodeBlocks  = (nNodes + 255) / 256;
    const int edgeBlocks  = (nEdges + 255) / 256;
    const int scatBlocks  = (int)(((long long)nEdges * 64 + 255) / 256);
    const int ndBlocks    = (int)((ND + 255) / 256);

    for (int g = 0; g < 2; ++g) {
        const int*   ei   = (const int*)d_in[g];
        const int*   src  = ei;
        const int*   dst  = ei + nEdges;
        const float* feat = (const float*)d_in[2 + g];
        const float* hid  = (const float*)d_in[4 + g];
        float*       outg = (float*)d_out + (size_t)g * ND;

        prep_kernel<<<prepBlocks, 256, 0, stream>>>(feat, hid, Ahi, Alo, (int)(ND / 2));
        gemm_kernel<<<gemmBlocks, 256, 0, stream>>>(Ahi, Alo, Whi, Wlo, H, nNodes);

        fill_kernel<<<nodeBlocks, 256, 0, stream>>>(deg, 1.0f, nNodes);   // self-loop
        deg_kernel<<<edgeBlocks, 256, 0, stream>>>(dst, deg, nEdges);
        dinv_kernel<<<nodeBlocks, 256, 0, stream>>>(deg, nNodes);

        fill_kernel<<<ndBlocks, 256, 0, stream>>>(outg, 0.0f, (int)ND);
        scatter_kernel<<<scatBlocks, 256, 0, stream>>>(src, dst, deg, H, outg, nEdges);
        epilogue_kernel<<<ndBlocks, 256, 0, stream>>>(H, deg, b, outg, (long long)ND);

        // pass-through hidden state
        hipMemcpyAsync((float*)d_out + (size_t)(2 + g) * ND, hid, ND * sizeof(float),
                       hipMemcpyDeviceToDevice, stream);
    }
}
